// TriFeaSlim_83408264888950
// MI455X (gfx1250) — matrix operand order repaired and checked by reading, not compiled, not run on hardware
//
#include <hip/hip_runtime.h>
#include <hip/hip_bf16.h>

// PointNet++-style pipeline for MI455X (gfx1250, wave32, WMMA).
// All 1x1-conv MLP layers run as f16xf16->f32 WMMA GEMMs (v_wmma_f32_16x16x32_f16)
// with BN+bias+ReLU folded into a per-channel alpha/beta epilogue.
// FPS/KNN are LDS-staged scalar kernels; grouping/3NN-interp are scalar (no LDS).

typedef _Float16 h16;
typedef __attribute__((ext_vector_type(16))) _Float16 v16h;
typedef __attribute__((ext_vector_type(8)))  _Float16 v8h;
typedef __attribute__((ext_vector_type(8)))  float    v8f;

#define BB 8
#define NN 1024
#define SS1 921
#define KK1 100
#define SS2 829
#define KK2 50

__host__ __device__ constexpr int pad16i(int x){ return (x + 15) & ~15; }
__host__ __device__ constexpr int pad32i(int x){ return (x + 31) & ~31; }

// ---------------------------------------------------------------------------
// Weight packing: W [Cin x Cout] f32 -> W^T [Npad x Kpad] f16 (zero padded),
// plus per-channel (alpha, beta): y = alpha*acc + beta, alpha = g*rsqrt(rv+eps),
// beta = (b - rm)*alpha + be  (or alpha=1, beta=b without BN).
// ---------------------------------------------------------------------------
__global__ void __launch_bounds__(256)
pack_layer(const float* __restrict__ w, const float* __restrict__ b,
           const float* __restrict__ g, const float* __restrict__ be,
           const float* __restrict__ rm, const float* __restrict__ rv,
           int cin, int cout, int Kpad, int Npad,
           h16* __restrict__ wt, float2* __restrict__ ab)
{
    int t = blockIdx.x * blockDim.x + threadIdx.x;
    if (t >= Npad * Kpad) return;
    int n = t / Kpad, k = t % Kpad;
    float v = (n < cout && k < cin) ? w[k * cout + n] : 0.f;
    wt[(size_t)n * Kpad + k] = (h16)v;
    if (k == 0) {
        if (n < cout) {
            float sc = 1.f, sh;
            if (g) { sc = g[n] * rsqrtf(rv[n] + 1e-5f); sh = (b[n] - rm[n]) * sc + be[n]; }
            else   { sh = b[n]; }
            ab[n] = make_float2(sc, sh);
        } else {
            ab[n] = make_float2(0.f, 0.f);
        }
    }
}

// ---------------------------------------------------------------------------
// WMMA GEMM: out[M x Cout] = A[M x Kpad](f16) * W^T[Npad x Kpad](f16), fused
// epilogue. One wave computes one 16x16 tile; K in steps of 32.
// A fragment (16-bit A 16x32, ISA 7.12.2): lane<16 -> K{k0..k0+7, k0+16..k0+23},
// lane>=16 -> K{k0+8..k0+15, k0+24..k0+31}; row M = tm + (lane&15).
// B fragment (32x16): lane l holds column N = tn + (l&15), same K split as A:
// lane<16 -> K{k0..k0+7, k0+16..k0+23}, lane>=16 -> K{k0+8..k0+15, k0+24..k0+31}.
// D: VGPR r -> row tm + r + 8*(lane>=16), col = tn + (lane&15).
// ---------------------------------------------------------------------------
__global__ void __launch_bounds__(256)
gemm_wmma(const h16* __restrict__ A, int lda,
          const h16* __restrict__ Wt, int ldw,
          const float2* __restrict__ ab,
          h16* __restrict__ outH, float* __restrict__ outF, int ldo,
          int M, int Npad, int Kpad, int Cout, int relu)
{
    int wave = blockIdx.x * (blockDim.x >> 5) + (threadIdx.x >> 5);
    int mt = M >> 4, nt = Npad >> 4;
    if (wave >= mt * nt) return;           // wave-uniform: EXEC stays all-ones
    int tm = (wave % mt) << 4;
    int tn = (wave / mt) << 4;
    int lane = threadIdx.x & 31;
    int hi = lane >> 4;
    int l  = lane & 15;

    const h16* arow = A  + (size_t)(tm + l) * lda;
    const h16* brow = Wt + (size_t)(tn + l) * ldw;
    v8f acc = {};
    for (int k0 = 0; k0 < Kpad; k0 += 32) {
        __builtin_prefetch(arow + k0 + 64, 0, 1);   // global_prefetch
        v8h alo = *(const v8h*)(arow + k0 + hi * 8);
        v8h ahi = *(const v8h*)(arow + k0 + 16 + hi * 8);
        v16h a;
#pragma unroll
        for (int i = 0; i < 8; ++i) { a[i] = alo[i]; a[8 + i] = ahi[i]; }
        v8h blo = *(const v8h*)(brow + k0 + hi * 8);
        v8h bhi = *(const v8h*)(brow + k0 + 16 + hi * 8);
        v16h bfrag;
#pragma unroll
        for (int i = 0; i < 8; ++i) { bfrag[i] = blo[i]; bfrag[8 + i] = bhi[i]; }
        acc = __builtin_amdgcn_wmma_f32_16x16x32_f16(
            false, a, false, bfrag, (short)0, acc, false, false);
    }

    int col = tn + l;
    float2 s = ab[col];
#pragma unroll
    for (int r = 0; r < 8; ++r) {
        int m = tm + r + hi * 8;
        float v = acc[r] * s.x + s.y;
        if (relu) v = fmaxf(v, 0.f);
        if (col >= Cout) v = 0.f;
        if (outH) outH[(size_t)m * ldo + col] = (h16)v;
        else if (col < Cout) outF[(size_t)m * ldo + col] = v;
    }
}

// ---------------------------------------------------------------------------
// Farthest point sampling. One block per batch, points staged in LDS.
// ---------------------------------------------------------------------------
__global__ void __launch_bounds__(256)
fps_kernel(const float* __restrict__ xyz, int n, int nc,
           int* __restrict__ idx_out, float* __restrict__ new_xyz)
{
    __shared__ float px[NN], py[NN], pz[NN], dist[NN];
    __shared__ float bestd[256];
    __shared__ int   besti[256];
    int b = blockIdx.x, t = threadIdx.x;
    for (int i = t; i < n; i += 256) {
        const float* p = xyz + (size_t)(b * n + i) * 3;
        px[i] = p[0]; py[i] = p[1]; pz[i] = p[2];
        dist[i] = 1e10f;
    }
    __syncthreads();
    int far = 0;
    for (int it = 0; it < nc; ++it) {
        if (t == 0) {
            idx_out[b * nc + it] = far;
            float* o = new_xyz + (size_t)(b * nc + it) * 3;
            o[0] = px[far]; o[1] = py[far]; o[2] = pz[far];
        }
        float cx = px[far], cy = py[far], cz = pz[far];
        float bd = -1.f; int bi = 0;
        for (int i = t; i < n; i += 256) {
            float dx = px[i] - cx, dy = py[i] - cy, dz = pz[i] - cz;
            float d = dx * dx + dy * dy + dz * dz;
            float nd = fminf(dist[i], d);
            dist[i] = nd;
            if (nd > bd) { bd = nd; bi = i; }
        }
        bestd[t] = bd; besti[t] = bi;
        __syncthreads();
        for (int off = 128; off > 0; off >>= 1) {
            if (t < off && bestd[t + off] > bestd[t]) {
                bestd[t] = bestd[t + off]; besti[t] = besti[t + off];
            }
            __syncthreads();
        }
        far = besti[0];
        __syncthreads();
    }
}

// ---------------------------------------------------------------------------
// k-NN indices (ascending by (dist, idx)); refs staged in LDS.
// grid: (ceil(nq/256), B)
// ---------------------------------------------------------------------------
__global__ void __launch_bounds__(256)
knn_idx(const float* __restrict__ q, const float* __restrict__ r,
        int nq, int nr, int k, int* __restrict__ idx)
{
    __shared__ float rx[NN], ry[NN], rz[NN];
    int b = blockIdx.y;
    for (int i = threadIdx.x; i < nr; i += 256) {
        const float* p = r + (size_t)(b * nr + i) * 3;
        rx[i] = p[0]; ry[i] = p[1]; rz[i] = p[2];
    }
    __syncthreads();
    int qi = blockIdx.x * 256 + threadIdx.x;
    if (qi >= nq) return;
    const float* qp = q + (size_t)(b * nq + qi) * 3;
    float qx = qp[0], qy = qp[1], qz = qp[2];
    float lastd = -1.f; int lasti = -1;
    int* out = idx + ((size_t)(b * nq + qi)) * k;
    for (int j = 0; j < k; ++j) {
        float bd = 1e30f; int bi = 0;
        for (int i = 0; i < nr; ++i) {
            float dx = rx[i] - qx, dy = ry[i] - qy, dz = rz[i] - qz;
            float d = dx * dx + dy * dy + dz * dz;
            bool gt = (d > lastd) || (d == lastd && i > lasti);
            if (gt && d < bd) { bd = d; bi = i; }
        }
        out[j] = bi;
        lastd = bd; lasti = bi;
    }
}

// ---------------------------------------------------------------------------
// Build grouped features: row (b,s,k) = [xyz[idx]-center (3), feats[idx] (C)],
// zero-padded to lda, written f16 (WMMA A staging).
// ---------------------------------------------------------------------------
__global__ void __launch_bounds__(256)
group_stage(const float* __restrict__ xyz, const float* __restrict__ new_xyz,
            const int* __restrict__ idx, const h16* __restrict__ feats,
            int ldf, int C, int S, int K, int nr,
            h16* __restrict__ A, int lda)
{
    long long r = (long long)blockIdx.x * 256 + threadIdx.x;
    long long total = (long long)BB * S * K;
    if (r >= total) return;
    int k = (int)(r % K);
    int s = (int)((r / K) % S);
    int b = (int)(r / ((long long)K * S));
    int id = idx[((size_t)(b * S + s)) * K + k];
    const float* pr = xyz     + (size_t)(b * nr + id) * 3;
    const float* pc = new_xyz + (size_t)(b * S  + s ) * 3;
    h16* arow = A + (size_t)r * lda;
    arow[0] = (h16)(pr[0] - pc[0]);
    arow[1] = (h16)(pr[1] - pc[1]);
    arow[2] = (h16)(pr[2] - pc[2]);
    const h16* f = feats + (size_t)(b * nr + id) * ldf;
    for (int c = 0; c < C; ++c) arow[3 + c] = f[c];
    for (int c = 3 + C; c < lda; ++c) arow[c] = (h16)0.f;
}

// Max-pool over the K neighborhood: [rowsOut*K x ld] -> [rowsOut x ldo]
__global__ void __launch_bounds__(256)
maxpool_k(const h16* __restrict__ X, int ld, int K, int C, int ldo,
          h16* __restrict__ out, long long rowsOut)
{
    long long t = (long long)blockIdx.x * 256 + threadIdx.x;
    if (t >= rowsOut * ldo) return;
    long long ro = t / ldo;
    int c = (int)(t % ldo);
    if (c >= C) { out[ro * ldo + c] = (h16)0.f; return; }
    const h16* p = X + ro * (size_t)K * ld + c;
    float m = -1e30f;
    for (int k = 0; k < K; ++k) m = fmaxf(m, (float)p[(size_t)k * ld]);
    out[ro * ldo + c] = (h16)m;
}

// ---------------------------------------------------------------------------
// Feature propagation staging: 3-NN inverse-distance interpolation fused with
// concat -> f16 A rows. feats1==nullptr means points1 = [xyz, xyz] (6 ch).
// ---------------------------------------------------------------------------
__global__ void __launch_bounds__(256)
interp_stage(const float* __restrict__ xyz1, const float* __restrict__ xyz2,
             const h16* __restrict__ feats1, int ldf1, int C1,
             const h16* __restrict__ feats2, int ldf2, int C2,
             int n1, int n2, h16* __restrict__ A, int lda, int Mpad)
{
    int r = blockIdx.x * 256 + threadIdx.x;
    if (r >= Mpad) return;
    h16* arow = A + (size_t)r * lda;
    if (r >= BB * n1) { for (int c = 0; c < lda; ++c) arow[c] = (h16)0.f; return; }
    int b = r / n1, q = r % n1;
    const float* p = xyz1 + (size_t)(b * n1 + q) * 3;
    float qx = p[0], qy = p[1], qz = p[2];
    float d0 = 1e30f, d1 = 1e30f, d2 = 1e30f;
    int i0 = 0, i1 = 0, i2 = 0;
    for (int i = 0; i < n2; ++i) {
        const float* rp = xyz2 + (size_t)(b * n2 + i) * 3;
        float dx = qx - rp[0], dy = qy - rp[1], dz = qz - rp[2];
        float d = dx * dx + dy * dy + dz * dz;
        if (d < d0)      { d2 = d1; i2 = i1; d1 = d0; i1 = i0; d0 = d; i0 = i; }
        else if (d < d1) { d2 = d1; i2 = i1; d1 = d; i1 = i; }
        else if (d < d2) { d2 = d; i2 = i; }
    }
    float w0 = 1.f / (d0 + 1e-8f), w1 = 1.f / (d1 + 1e-8f), w2 = 1.f / (d2 + 1e-8f);
    float ws = w0 + w1 + w2; w0 /= ws; w1 /= ws; w2 /= ws;
    int c = 0;
    if (feats1) {
        const h16* f1 = feats1 + (size_t)(b * n1 + q) * ldf1;
        for (; c < C1; ++c) arow[c] = f1[c];
    } else {
        arow[0] = (h16)qx; arow[1] = (h16)qy; arow[2] = (h16)qz;
        arow[3] = (h16)qx; arow[4] = (h16)qy; arow[5] = (h16)qz;
        c = 6;
    }
    const h16* fa = feats2 + (size_t)(b * n2 + i0) * ldf2;
    const h16* fb = feats2 + (size_t)(b * n2 + i1) * ldf2;
    const h16* fc = feats2 + (size_t)(b * n2 + i2) * ldf2;
    for (int j = 0; j < C2; ++j)
        arow[c + j] = (h16)(w0 * (float)fa[j] + w1 * (float)fb[j] + w2 * (float)fc[j]);
    for (int j = c + C2; j < lda; ++j) arow[j] = (h16)0.f;
}

// xyz (f32, 3ch) -> f16 rows with ld=4 (padded)
__global__ void __launch_bounds__(256)
xyz_to_h(const float* __restrict__ xyz, h16* __restrict__ out, int n)
{
    int t = blockIdx.x * 256 + threadIdx.x;
    if (t >= BB * n * 4) return;
    int c = t & 3, r = t >> 2;
    out[t] = (c < 3) ? (h16)xyz[(size_t)r * 3 + c] : (h16)0.f;
}

__global__ void __launch_bounds__(256)
lsm_kernel(const float* __restrict__ X, int ld, int C, int n, float* __restrict__ out)
{
    int i = blockIdx.x * 256 + threadIdx.x;
    if (i >= n) return;
    const float* x = X + (size_t)i * ld;
    float m = -1e30f;
    for (int c = 0; c < C; ++c) m = fmaxf(m, x[c]);
    float s = 0.f;
    for (int c = 0; c < C; ++c) s += expf(x[c] - m);
    float l = logf(s);
    for (int c = 0; c < C; ++c) out[(size_t)i * C + c] = x[c] - m - l;
}

// ---------------------------------------------------------------------------
extern "C" void kernel_launch(void* const* d_in, const int* in_sizes, int n_in,
                              void* d_out, int out_size, void* d_ws, size_t ws_size,
                              hipStream_t stream)
{
    (void)in_sizes; (void)out_size; (void)ws_size;
    const int M1  = BB * SS1 * KK1;   // 736800 (mult of 16)
    const int M2  = BB * SS2 * KK2;   // 331600 (mult of 16)
    const int MF2 = pad16i(BB * SS1); // 7376
    const int MPT = BB * NN;          // 8192

    // -------- parameter table (setup_inputs insertion order) --------
    int cur = 0;
    auto nx = [&]() { return (const float*)d_in[cur++ < n_in ? cur - 1 : n_in - 1]; };
    const float* xyz = nx();
    struct L { const float *w, *b, *g, *be, *rm, *rv; int cin, cout, relu; };
    L ls[23]; int nl = 0;
    auto addbn = [&](int ci, int co) {
        L x; x.w = nx(); x.b = nx(); x.g = nx(); x.be = nx(); x.rm = nx(); x.rv = nx();
        x.cin = ci; x.cout = co; x.relu = 1; ls[nl++] = x; };
    auto addcv = [&](int ci, int co) {
        L x; x.w = nx(); x.b = nx(); x.g = nullptr; x.be = nullptr; x.rm = nullptr; x.rv = nullptr;
        x.cin = ci; x.cout = co; x.relu = 0; ls[nl++] = x; };
    addbn(6, 8);    addbn(8, 16);   addbn(16, 32);    // 0-2  sa1
    addbn(35, 64);  addbn(64, 128); addbn(128, 256);  // 3-5  sa2
    addbn(288, 256);addbn(256, 128);addbn(128, 64);   // 6-8  fp2
    addbn(70, 128); addbn(128, 64); addbn(64, 32);    // 9-11 fp1
    addbn(32, 64);                                     // 12   conv1+bn1
    addcv(64, 128);                                    // 13   conv2
    addbn(128, 128); addbn(128, 32); addcv(32, 3);     // 14-16 eula
    addbn(128, 128); addbn(128, 32); addcv(32, 2);     // 17-19 edge
    addbn(128, 128); addbn(128, 64); addcv(64, 4);     // 20-22 meta

    // -------- workspace bump allocator --------
    char* wsb = (char*)d_ws;
    size_t off = 0;
    auto alloc = [&](size_t bytes) -> char* {
        off = (off + 255) & ~(size_t)255;
        char* p = wsb + off; off += bytes; return p; };

    const size_t PBYTES = (size_t)M2 * 256 * sizeof(h16);   // largest staged act
    h16*  P0      = (h16*)alloc(PBYTES);
    h16*  P1      = (h16*)alloc(PBYTES);
    h16*  xyz16   = (h16*)alloc((size_t)MPT * 4 * sizeof(h16));
    int*  idxf1   = (int*)alloc((size_t)BB * SS1 * sizeof(int));
    float* nxyz1  = (float*)alloc((size_t)BB * SS1 * 3 * sizeof(float));
    int*  idx1    = (int*)alloc((size_t)BB * SS1 * KK1 * sizeof(int));
    h16*  pooled1 = (h16*)alloc((size_t)BB * SS1 * 32 * sizeof(h16));
    int*  idxf2   = (int*)alloc((size_t)BB * SS2 * sizeof(int));
    float* nxyz2  = (float*)alloc((size_t)BB * SS2 * 3 * sizeof(float));
    int*  idx2    = (int*)alloc((size_t)BB * SS2 * KK2 * sizeof(int));
    h16*  pooled2 = (h16*)alloc((size_t)BB * SS2 * 256 * sizeof(h16));
    h16*  l1f     = (h16*)alloc((size_t)MF2 * 64 * sizeof(h16));
    h16*  l0f     = (h16*)alloc((size_t)MPT * 32 * sizeof(h16));
    h16*  feat2   = (h16*)alloc((size_t)MPT * 128 * sizeof(h16));
    float* hstage = (float*)alloc((size_t)MPT * 4 * sizeof(float));
    h16*   wt[23];
    float2* abp[23];
    for (int i = 0; i < nl; ++i) {
        int Kp = pad32i(ls[i].cin), Np = pad32i(ls[i].cout);
        wt[i]  = (h16*)alloc((size_t)Np * Kp * sizeof(h16));
        abp[i] = (float2*)alloc((size_t)Np * sizeof(float2));
    }

    // -------- pack weights + fused BN epilogues --------
    for (int i = 0; i < nl; ++i) {
        int Kp = pad32i(ls[i].cin), Np = pad32i(ls[i].cout);
        int tot = Np * Kp;
        pack_layer<<<(tot + 255) / 256, 256, 0, stream>>>(
            ls[i].w, ls[i].b, ls[i].g, ls[i].be, ls[i].rm, ls[i].rv,
            ls[i].cin, ls[i].cout, Kp, Np, wt[i], abp[i]);
    }

    auto gemm = [&](int li, const h16* A, int lda, h16* oh, float* of, int ldo, int M) {
        int Kp = pad32i(ls[li].cin), Np = pad32i(ls[li].cout);
        int tiles = (M / 16) * (Np / 16);
        gemm_wmma<<<(tiles + 7) / 8, 256, 0, stream>>>(
            A, lda, wt[li], Kp, abp[li], oh, of, ldo, M, Np, Kp, ls[li].cout, ls[li].relu);
    };

    // -------- SA1 --------
    xyz_to_h<<<(BB * NN * 4 + 255) / 256, 256, 0, stream>>>(xyz, xyz16, NN);
    fps_kernel<<<BB, 256, 0, stream>>>(xyz, NN, SS1, idxf1, nxyz1);
    knn_idx<<<dim3((SS1 + 255) / 256, BB), 256, 0, stream>>>(nxyz1, xyz, SS1, NN, KK1, idx1);
    group_stage<<<(int)(((long long)M1 + 255) / 256), 256, 0, stream>>>(
        xyz, nxyz1, idx1, xyz16, 4, 3, SS1, KK1, NN, P0, 32);
    gemm(0, P0, 32, P1, nullptr, 32, M1);
    gemm(1, P1, 32, P0, nullptr, 32, M1);
    gemm(2, P0, 32, P1, nullptr, 32, M1);
    maxpool_k<<<(int)(((long long)BB * SS1 * 32 + 255) / 256), 256, 0, stream>>>(
        P1, 32, KK1, 32, 32, pooled1, (long long)BB * SS1);

    // -------- SA2 --------
    fps_kernel<<<BB, 256, 0, stream>>>(nxyz1, SS1, SS2, idxf2, nxyz2);
    knn_idx<<<dim3((SS2 + 255) / 256, BB), 256, 0, stream>>>(nxyz2, nxyz1, SS2, SS1, KK2, idx2);
    group_stage<<<(int)(((long long)M2 + 255) / 256), 256, 0, stream>>>(
        nxyz1, nxyz2, idx2, pooled1, 32, 32, SS2, KK2, SS1, P0, 64);
    gemm(3, P0, 64, P1, nullptr, 64, M2);
    gemm(4, P1, 64, P0, nullptr, 128, M2);
    gemm(5, P0, 128, P1, nullptr, 256, M2);
    maxpool_k<<<(int)(((long long)BB * SS2 * 256 + 255) / 256), 256, 0, stream>>>(
        P1, 256, KK2, 256, 256, pooled2, (long long)BB * SS2);

    // -------- FP2 --------
    interp_stage<<<(MF2 + 255) / 256, 256, 0, stream>>>(
        nxyz1, nxyz2, pooled1, 32, 32, pooled2, 256, 256, SS1, SS2, P0, 288, MF2);
    gemm(6, P0, 288, P1, nullptr, 256, MF2);
    gemm(7, P1, 256, P0, nullptr, 128, MF2);
    gemm(8, P0, 128, l1f, nullptr, 64, MF2);

    // -------- FP1 --------
    interp_stage<<<(MPT + 255) / 256, 256, 0, stream>>>(
        xyz, nxyz1, nullptr, 0, 6, l1f, 64, 64, NN, SS1, P0, 96, MPT);
    gemm(9,  P0, 96,  P1, nullptr, 128, MPT);
    gemm(10, P1, 128, P0, nullptr, 64,  MPT);
    gemm(11, P0, 64,  l0f, nullptr, 32, MPT);

    // -------- trunk + heads --------
    gemm(12, l0f, 32, P0, nullptr, 64,  MPT);   // conv1 + bn1 + relu
    gemm(13, P0, 64,  feat2, nullptr, 128, MPT);// conv2 (raw)

    float* out = (float*)d_out;
    // eula: identity activation, write f32 compact [8192 x 3]
    gemm(14, feat2, 128, P0, nullptr, 128, MPT);
    gemm(15, P0, 128, P1, nullptr, 32, MPT);
    gemm(16, P1, 32, nullptr, out, 3, MPT);
    // edge: log_softmax over 2
    gemm(17, feat2, 128, P0, nullptr, 128, MPT);
    gemm(18, P0, 128, P1, nullptr, 32, MPT);
    gemm(19, P1, 32, nullptr, hstage, 2, MPT);
    lsm_kernel<<<(MPT + 255) / 256, 256, 0, stream>>>(hstage, 2, 2, MPT, out + 24576);
    // meta: log_softmax over 4
    gemm(20, feat2, 128, P0, nullptr, 128, MPT);
    gemm(21, P0, 128, P1, nullptr, 64, MPT);
    gemm(22, P1, 64, nullptr, hstage, 4, MPT);
    lsm_kernel<<<(MPT + 255) / 256, 256, 0, stream>>>(hstage, 4, 4, MPT, out + 40960);
}
